// DshmmLm_22531398435440
// MI455X (gfx1250) — matrix operand order repaired and checked by reading, not compiled, not run on hardware
//
#include <hip/hip_runtime.h>
#include <hip/hip_bf16.h>
#include <math.h>

#define VV 32000
#define CC 4096
#define SS 32
#define HH 512
#define BB 32
#define TT 256

typedef __bf16 bf16_t;
typedef __attribute__((ext_vector_type(16))) __bf16 v16bf;
typedef __attribute__((ext_vector_type(8)))  __bf16 v8bf;
typedef __attribute__((ext_vector_type(8)))  float  v8f;

union FragBF { v16bf v; v8bf h[2]; };

// ---------------------------------------------------------------------------
// CAS-based float atomic max
// ---------------------------------------------------------------------------
__device__ inline void atomicMaxF(float* addr, float v) {
    unsigned int* ai = (unsigned int*)addr;
    unsigned int old = *ai;
    while (true) {
        float of = __uint_as_float(old);
        if (of >= v) break;
        unsigned int prev = atomicCAS(ai, old, __float_as_uint(v));
        if (prev == old) break;
        old = prev;
    }
}

// ---------------------------------------------------------------------------
// fp32 -> bf16 conversion
// ---------------------------------------------------------------------------
__global__ void f2bf_kernel(const float* __restrict__ in, bf16_t* __restrict__ out, int n) {
    int i = blockIdx.x * 256 + threadIdx.x;
    if (i < n) out[i] = (bf16_t)in[i];
}

// ---------------------------------------------------------------------------
// init: zero output scalar, emission rowmax=-inf/rowsum=0, trans rowmax=-inf
// ---------------------------------------------------------------------------
__global__ void init_kernel(float* out, float* rowmax_e, float* rowsum_e, float* rowmax_t) {
    int i = blockIdx.x * 256 + threadIdx.x;
    if (i == 0) out[0] = 0.0f;
    if (i < CC) { rowmax_e[i] = -INFINITY; rowsum_e[i] = 0.0f; rowmax_t[i] = -INFINITY; }
}

// ---------------------------------------------------------------------------
// WMMA GEMM (K fixed = 512): OUT[M,N] = epi( A[M,K](bf16) @ B[N,K]^T(bf16) )
// One wave computes a 32x64 strip: 2 A fragments x 4 B fragments -> 8 WMMAs
// per k-step (12 loads / 8 WMMAs), all loads issued into distinct registers
// before the WMMA burst so the scheduler can use partial s_wait_loadcnt.
// flags: 1 = +bias[n], 2 = relu, 4 = +resid[m,n] (fp32),
//        8 = fused per-row atomic max into rowmax_out (excluding col N-1)
// 16-bit A fragment as loaded here (contiguous k, not the ISA table order): lane L holds row M=L&15;
//   elements e -> K = kb+e, kb=(L>=16)?16:0.
// 16-bit B fragment: lane L holds col N=L&15; elements e -> K=(L>=16?16:0)+e.
// C/D: lane L -> N=L&15, VGPR r -> M = r + (L>=16?8:0).
// ---------------------------------------------------------------------------
__global__ __launch_bounds__(32) void wmma_gemm_kernel(
    const bf16_t* __restrict__ A, const bf16_t* __restrict__ Bm,
    const float* __restrict__ bias, const float* __restrict__ resid,
    float* __restrict__ outF, float* __restrict__ rowmax_out,
    int M, int N, int flags)
{
    const int lane = threadIdx.x;
    const int row0 = blockIdx.x * 32;
    const int col0 = blockIdx.y * 64;
    const int mA   = lane & 15;
    const int kbA  = (lane >= 16) ? 16 : 0;
    const int kbB  = (lane >= 16) ? 16 : 0;

    const bf16_t* arow0 = A + (size_t)(row0 + mA) * HH + kbA;
    const bf16_t* arow1 = arow0 + (size_t)16 * HH;
    const bf16_t* brow0 = Bm + (size_t)(col0 + mA) * HH + kbB;
    const bf16_t* brow1 = brow0 + (size_t)16 * HH;
    const bf16_t* brow2 = brow0 + (size_t)32 * HH;
    const bf16_t* brow3 = brow0 + (size_t)48 * HH;

    v8f zero = {0.f,0.f,0.f,0.f,0.f,0.f,0.f,0.f};
    v8f acc[2][4] = {{zero, zero, zero, zero}, {zero, zero, zero, zero}};

    for (int kk = 0; kk < HH; kk += 32) {
        // prefetch next k-block (global_prefetch_b8; speculative, safe at edges)
        __builtin_prefetch(arow0 + kk + 64, 0, 1);
        __builtin_prefetch(brow0 + kk + 64, 0, 1);

        FragBF a0, a1, b0, b1, b2, b3;
        a0.h[0] = *(const v8bf*)(arow0 + kk);
        a0.h[1] = *(const v8bf*)(arow0 + kk + 8);
        a1.h[0] = *(const v8bf*)(arow1 + kk);
        a1.h[1] = *(const v8bf*)(arow1 + kk + 8);
        b0.h[0] = *(const v8bf*)(brow0 + kk);
        b0.h[1] = *(const v8bf*)(brow0 + kk + 8);
        b1.h[0] = *(const v8bf*)(brow1 + kk);
        b1.h[1] = *(const v8bf*)(brow1 + kk + 8);
        b2.h[0] = *(const v8bf*)(brow2 + kk);
        b2.h[1] = *(const v8bf*)(brow2 + kk + 8);
        b3.h[0] = *(const v8bf*)(brow3 + kk);
        b3.h[1] = *(const v8bf*)(brow3 + kk + 8);

        acc[0][0] = __builtin_amdgcn_wmma_f32_16x16x32_bf16(false, a0.v, false, b0.v, (short)0, acc[0][0], false, false);
        acc[0][1] = __builtin_amdgcn_wmma_f32_16x16x32_bf16(false, a0.v, false, b1.v, (short)0, acc[0][1], false, false);
        acc[0][2] = __builtin_amdgcn_wmma_f32_16x16x32_bf16(false, a0.v, false, b2.v, (short)0, acc[0][2], false, false);
        acc[0][3] = __builtin_amdgcn_wmma_f32_16x16x32_bf16(false, a0.v, false, b3.v, (short)0, acc[0][3], false, false);
        acc[1][0] = __builtin_amdgcn_wmma_f32_16x16x32_bf16(false, a1.v, false, b0.v, (short)0, acc[1][0], false, false);
        acc[1][1] = __builtin_amdgcn_wmma_f32_16x16x32_bf16(false, a1.v, false, b1.v, (short)0, acc[1][1], false, false);
        acc[1][2] = __builtin_amdgcn_wmma_f32_16x16x32_bf16(false, a1.v, false, b2.v, (short)0, acc[1][2], false, false);
        acc[1][3] = __builtin_amdgcn_wmma_f32_16x16x32_bf16(false, a1.v, false, b3.v, (short)0, acc[1][3], false, false);
    }

    const int nL = lane & 15;
    const int mo = (lane >= 16) ? 8 : 0;
#pragma unroll
    for (int mi = 0; mi < 2; ++mi) {
#pragma unroll
        for (int j = 0; j < 4; ++j) {
#pragma unroll
            for (int r = 0; r < 8; ++r) {
                int m = row0 + 16 * mi + mo + r;
                int n = col0 + 16 * j + nL;
                float x = acc[mi][j][r];
                if (flags & 1) x += bias[n];
                if (flags & 4) x += resid[(size_t)m * N + n];
                if (flags & 2) x = fmaxf(x, 0.0f);
                outF[(size_t)m * N + n] = x;
            }
        }
        if (flags & 8) {
            // fused per-row max (excluding column N-1, which is -inf in the ref)
#pragma unroll
            for (int r = 0; r < 8; ++r) {
                float rm = -INFINITY;
#pragma unroll
                for (int j = 0; j < 4; ++j) {
                    int n = col0 + 16 * j + nL;
                    float x = acc[mi][j][r];
                    if (n != N - 1) rm = fmaxf(rm, x);
                }
                // reduce across the 16-lane half (xor<16 stays within the half)
                for (int offx = 8; offx > 0; offx >>= 1)
                    rm = fmaxf(rm, __shfl_xor(rm, offx, 32));
                if (nL == 0) atomicMaxF(&rowmax_out[row0 + 16 * mi + mo + r], rm);
            }
        }
    }
}

// ---------------------------------------------------------------------------
// Row log-sum-exp over ncols_used leading columns (computes its own max)
// ---------------------------------------------------------------------------
__global__ void row_lse_kernel(const float* __restrict__ x, int ncols_used, int ld,
                               float* __restrict__ out)
{
    __shared__ float sm[256];
    int row = blockIdx.x;
    const float* xr = x + (size_t)row * ld;
    float m = -INFINITY;
    for (int i = threadIdx.x; i < ncols_used; i += 256) m = fmaxf(m, xr[i]);
    sm[threadIdx.x] = m; __syncthreads();
    for (int s = 128; s > 0; s >>= 1) {
        if (threadIdx.x < s) sm[threadIdx.x] = fmaxf(sm[threadIdx.x], sm[threadIdx.x + s]);
        __syncthreads();
    }
    float mm = sm[0]; __syncthreads();
    float a = 0.0f;
    for (int i = threadIdx.x; i < ncols_used; i += 256) a += __expf(xr[i] - mm);
    sm[threadIdx.x] = a; __syncthreads();
    for (int s = 128; s > 0; s >>= 1) {
        if (threadIdx.x < s) sm[threadIdx.x] += sm[threadIdx.x + s];
        __syncthreads();
    }
    if (threadIdx.x == 0) out[row] = mm + __logf(sm[0]);
}

// ---------------------------------------------------------------------------
// Row log-sum-exp with precomputed per-row max (single pass over the matrix)
// ---------------------------------------------------------------------------
__global__ void row_sum_lse_kernel(const float* __restrict__ x, const float* __restrict__ rowmax,
                                   int ncols_used, int ld, float* __restrict__ out)
{
    __shared__ float sm[256];
    int row = blockIdx.x;
    float mm = rowmax[row];
    const float* xr = x + (size_t)row * ld;
    float a = 0.0f;
    for (int i = threadIdx.x; i < ncols_used; i += 256) a += __expf(xr[i] - mm);
    sm[threadIdx.x] = a; __syncthreads();
    for (int s = 128; s > 0; s >>= 1) {
        if (threadIdx.x < s) sm[threadIdx.x] += sm[threadIdx.x + s];
        __syncthreads();
    }
    if (threadIdx.x == 0) out[row] = mm + __logf(sm[0]);
}

// ---------------------------------------------------------------------------
// start head: score[c] = dot(h[c,:], w3) + b3[0]   (wave per row)
// ---------------------------------------------------------------------------
__global__ void start_score_kernel(const float* __restrict__ h, const float* __restrict__ w3,
                                   const float* __restrict__ b3, float* __restrict__ score)
{
    int wave = threadIdx.x >> 5;
    int lane = threadIdx.x & 31;
    int row  = blockIdx.x * 8 + wave;
    const float* hr = h + (size_t)row * HH;
    float acc = 0.0f;
    for (int k = lane; k < HH; k += 32) acc += hr[k] * w3[k];
    for (int off = 16; off > 0; off >>= 1) acc += __shfl_xor(acc, off, 32);
    if (lane == 0) score[row] = acc + b3[0];
}

// ---------------------------------------------------------------------------
// Masked emission logits: for each (word v, slot s), c = w2s[v,s]:
//   el[v,s] = dot(h_term[c,:], term_w3[v,:]) + term_b3[v]
// also atomic-max into per-state rowmax. Block = word, 8 waves x 4 slots.
// ---------------------------------------------------------------------------
__global__ void el_kernel(const int* __restrict__ w2s, const float* __restrict__ hterm,
                          const float* __restrict__ w3, const float* __restrict__ b3,
                          float* __restrict__ elvs, float* __restrict__ rowmax)
{
    int v    = blockIdx.x;
    int wave = threadIdx.x >> 5;
    int lane = threadIdx.x & 31;
    const float* wr = w3 + (size_t)v * HH;
#pragma unroll
    for (int q = 0; q < 4; ++q) {
        int s = wave * 4 + q;
        int c = w2s[v * SS + s];
        const float* hr = hterm + (size_t)c * HH;
        float acc = 0.0f;
        for (int k = lane; k < HH; k += 32) acc += hr[k] * wr[k];
        for (int off = 16; off > 0; off >>= 1) acc += __shfl_xor(acc, off, 32);
        if (lane == 0) {
            float e = acc + b3[v];
            elvs[v * SS + s] = e;
            atomicMaxF(&rowmax[c], e);
        }
    }
}

// ---------------------------------------------------------------------------
// Emission denominator: sum exp(el - rowmax[c]) per state, de-duplicating
// repeated states within a word (mask is boolean, dup (c,v) counts once).
// ---------------------------------------------------------------------------
__global__ void emit_sum_kernel(const int* __restrict__ w2s, const float* __restrict__ elvs,
                                const float* __restrict__ rowmax, float* __restrict__ rowsum)
{
    int idx = blockIdx.x * 256 + threadIdx.x;
    if (idx >= VV * SS) return;
    int v = idx / SS, s = idx % SS;
    int c = w2s[idx];
    bool dup = false;
    for (int s2 = 0; s2 < s; ++s2)
        if (w2s[v * SS + s2] == c) { dup = true; break; }
    if (!dup) atomicAdd(&rowsum[c], __expf(elvs[idx] - rowmax[c]));
}

__global__ void emit_fin_kernel(const float* __restrict__ rowmax, const float* __restrict__ rowsum,
                                float* __restrict__ rowlse)
{
    int c = blockIdx.x * 256 + threadIdx.x;
    if (c < CC) rowlse[c] = (rowsum[c] > 0.0f) ? (rowmax[c] + __logf(rowsum[c])) : -INFINITY;
}

// ---------------------------------------------------------------------------
// HMM forward: one wave per batch element, lane j = slot state (S == 32 lanes)
// ---------------------------------------------------------------------------
__global__ __launch_bounds__(32) void forward_kernel(
    const int* __restrict__ text, const int* __restrict__ w2s,
    const float* __restrict__ elvs, const float* __restrict__ rowlse_e,
    const float* __restrict__ score_start, const float* __restrict__ slse,
    const float* __restrict__ tl, const float* __restrict__ rowlse_t,
    float* __restrict__ out)
{
    int b = blockIdx.x;
    int j = threadIdx.x;

    int v0 = text[b * TT + 0];
    int c  = w2s[v0 * SS + j];
    float obs0  = elvs[v0 * SS + j] - rowlse_e[c];
    float alpha = (score_start[c] - slse[0]) + obs0;   // init + obs(t=0)
    int cprev = c;

    for (int t = 1; t < TT; ++t) {
        int v  = text[b * TT + t];
        int cj = w2s[v * SS + j];
        float obs = elvs[v * SS + j] - rowlse_e[cj];
        // alpha_j(t) = obs + logsumexp_i( tl[ci,cj] - rowlse_t[ci] + alpha_i )
        float m = -INFINITY, ssum = 0.0f;
#pragma unroll 4
        for (int i = 0; i < 32; ++i) {
            int   ci = __shfl(cprev, i, 32);
            float ai = __shfl(alpha, i, 32);
            float x  = tl[(size_t)ci * CC + cj] - rowlse_t[ci] + ai;
            if (x > m) { ssum = ssum * __expf(m - x) + 1.0f; m = x; }
            else       { ssum += __expf(x - m); }
        }
        alpha = obs + m + __logf(ssum);
        cprev = cj;
    }

    float m = alpha;
    for (int off = 16; off > 0; off >>= 1) m = fmaxf(m, __shfl_xor(m, off, 32));
    float e = __expf(alpha - m);
    for (int off = 16; off > 0; off >>= 1) e += __shfl_xor(e, off, 32);
    if (j == 0) atomicAdd(out, m + __logf(e));
}

// ---------------------------------------------------------------------------
// host launcher
// ---------------------------------------------------------------------------
extern "C" void kernel_launch(void* const* d_in, const int* in_sizes, int n_in,
                              void* d_out, int out_size, void* d_ws, size_t ws_size,
                              hipStream_t stream)
{
    (void)in_sizes; (void)n_in; (void)out_size;

    const int*   text      = (const int*)d_in[0];
    const int*   w2s       = (const int*)d_in[1];
    const float* start_emb = (const float*)d_in[2];
    const float* start_w1  = (const float*)d_in[3];
    const float* start_b1  = (const float*)d_in[4];
    const float* start_w2  = (const float*)d_in[5];
    const float* start_b2  = (const float*)d_in[6];
    const float* start_w3  = (const float*)d_in[7];
    const float* start_b3  = (const float*)d_in[8];
    const float* state_emb = (const float*)d_in[9];
    const float* trans_w1  = (const float*)d_in[10];
    const float* trans_b1  = (const float*)d_in[11];
    const float* trans_w2  = (const float*)d_in[12];
    const float* trans_b2  = (const float*)d_in[13];
    const float* nsp       = (const float*)d_in[14];
    const float* pre_emb   = (const float*)d_in[15];
    const float* term_w1   = (const float*)d_in[16];
    const float* term_b1   = (const float*)d_in[17];
    const float* term_w2   = (const float*)d_in[18];
    const float* term_b2   = (const float*)d_in[19];
    const float* term_w3   = (const float*)d_in[20];
    const float* term_b3   = (const float*)d_in[21];
    float* out = (float*)d_out;

    char* ws = (char*)d_ws;
    size_t off = 0;
    auto alloc = [&](size_t bytes) -> void* {
        void* p = ws + off;
        off = (off + bytes + 255) & ~(size_t)255;
        return p;
    };
    float*  tl    = (float*) alloc((size_t)CC * CC * 4);   // 67.1 MB transition logits
    float*  h1f   = (float*) alloc((size_t)CC * HH * 4);
    float*  h2f   = (float*) alloc((size_t)CC * HH * 4);
    bf16_t* bufA  = (bf16_t*)alloc((size_t)CC * HH * 2);
    bf16_t* bufB  = (bf16_t*)alloc((size_t)CC * HH * 2);
    bf16_t* w1b   = (bf16_t*)alloc((size_t)HH * HH * 2);
    bf16_t* w2b   = (bf16_t*)alloc((size_t)HH * HH * 2);
    bf16_t* nspb  = (bf16_t*)alloc((size_t)CC * HH * 2);
    float*  elvs  = (float*) alloc((size_t)VV * SS * 4);   // 4.1 MB masked emission logits
    float*  score = (float*) alloc((size_t)CC * 4);
    float*  slse  = (float*) alloc(256);
    float*  rmt   = (float*) alloc((size_t)CC * 4);        // trans row max (fused in GEMM)
    float*  rlt   = (float*) alloc((size_t)CC * 4);        // trans row lse
    float*  rme   = (float*) alloc((size_t)CC * 4);        // emit row max
    float*  rse   = (float*) alloc((size_t)CC * 4);        // emit row sum
    float*  rle   = (float*) alloc((size_t)CC * 4);        // emit row lse
    if (off > ws_size) return;  // workspace too small; deterministic no-op

    auto conv = [&](const float* src, bf16_t* dst, size_t n) {
        f2bf_kernel<<<dim3((unsigned)((n + 255) / 256)), dim3(256), 0, stream>>>(src, dst, (int)n);
    };
    auto gemm = [&](const bf16_t* A, const bf16_t* Bm, const float* bias,
                    const float* resid, float* o, float* rmax, int M, int N, int flags) {
        wmma_gemm_kernel<<<dim3(M / 32, N / 64), dim3(32), 0, stream>>>(A, Bm, bias, resid, o, rmax, M, N, flags);
    };

    init_kernel<<<dim3(CC / 256), dim3(256), 0, stream>>>(out, rme, rse, rmt);

    // ---- start MLP -> score -> lse ----
    conv(start_emb, bufA, (size_t)CC * HH);
    conv(start_w1,  w1b,  (size_t)HH * HH);
    conv(start_w2,  w2b,  (size_t)HH * HH);
    gemm(bufA, w1b, start_b1, nullptr,   h1f, nullptr, CC, HH, 1 | 2);
    conv(h1f, bufB, (size_t)CC * HH);
    gemm(bufB, w2b, start_b2, start_emb, h2f, nullptr, CC, HH, 1 | 2 | 4);
    start_score_kernel<<<dim3(CC / 8), dim3(256), 0, stream>>>(h2f, start_w3, start_b3, score);
    row_lse_kernel<<<dim3(1), dim3(256), 0, stream>>>(score, CC, CC, slse);

    // ---- state MLP -> transition logits (WMMA 4096x4096x512, fused row max) ----
    conv(state_emb, bufA, (size_t)CC * HH);
    conv(trans_w1,  w1b,  (size_t)HH * HH);
    conv(trans_w2,  w2b,  (size_t)HH * HH);
    conv(nsp,       nspb, (size_t)CC * HH);
    gemm(bufA, w1b, trans_b1, nullptr,   h1f, nullptr, CC, HH, 1 | 2);
    conv(h1f, bufB, (size_t)CC * HH);
    gemm(bufB, w2b, trans_b2, state_emb, h2f, nullptr, CC, HH, 1 | 2 | 4);
    conv(h2f, bufA, (size_t)CC * HH);
    gemm(bufA, nspb, nullptr, nullptr, tl, rmt, CC, CC, 8);
    // single pass: lse = rowmax + log(sum exp(x - rowmax)) over cols 0..C-2
    row_sum_lse_kernel<<<dim3(CC), dim3(256), 0, stream>>>(tl, rmt, CC - 1, CC, rlt);

    // ---- term MLP -> masked emission logits + per-state lse ----
    conv(pre_emb, bufA, (size_t)CC * HH);
    conv(term_w1, w1b,  (size_t)HH * HH);
    conv(term_w2, w2b,  (size_t)HH * HH);
    gemm(bufA, w1b, term_b1, nullptr, h1f, nullptr, CC, HH, 1 | 2);
    conv(h1f, bufB, (size_t)CC * HH);
    gemm(bufB, w2b, term_b2, pre_emb, h2f, nullptr, CC, HH, 1 | 2 | 4);
    el_kernel<<<dim3(VV), dim3(256), 0, stream>>>(w2s, h2f, term_w3, term_b3, elvs, rme);
    emit_sum_kernel<<<dim3((VV * SS) / 256), dim3(256), 0, stream>>>(w2s, elvs, rme, rse);
    emit_fin_kernel<<<dim3(CC / 256), dim3(256), 0, stream>>>(rme, rse, rle);

    // ---- HMM forward recursion ----
    forward_kernel<<<dim3(BB), dim3(32), 0, stream>>>(text, w2s, elvs, rle, score, slse, tl, rlt, out);
}
